// AttentionRNN_89086211653813
// MI455X (gfx1250) — matrix operand order repaired and checked by reading, not compiled, not run on hardware
//
#include <hip/hip_runtime.h>
#include <math.h>

// ---------------------------------------------------------------------------
// AttentionRNN for MI455X (gfx1250, wave32, WMMA).
//   B=128, T=512, H=1024, V=128.  All GEMMs via v_wmma_f32_16x16x32_bf16.
// ---------------------------------------------------------------------------

#define VOCABN 128
#define HIDN   1024
#define BN     128
#define TN     512

typedef __attribute__((ext_vector_type(16))) __bf16      v16bf;
typedef __attribute__((ext_vector_type(8)))  float       v8f;
typedef __attribute__((ext_vector_type(8)))  unsigned int v8u;

__device__ __forceinline__ unsigned short f2bf(float f) {
  union { float f; unsigned u; } x; x.f = f;
  unsigned u = x.u;
  return (unsigned short)((u + 0x7FFFu + ((u >> 16) & 1u)) >> 16); // RNE
}

// Branchless tanh: 1 - 2/(exp(2x)+1).  exp(2x)->inf gives +1, ->0 gives -1,
// so saturation is exact; avoids ocml tanhf's divergent EXEC-mask path.
__device__ __forceinline__ float fast_tanh(float v) {
  float e2 = __expf(2.0f * v);
  return 1.0f - 2.0f * __builtin_amdgcn_rcpf(e2 + 1.0f);
}

__device__ __forceinline__ v8f wmma_bf16(v16bf a, v16bf b, v8f c) {
  // (neg_a, A, neg_b, B, c_mod, C, reuse_a, reuse_b)
  return __builtin_amdgcn_wmma_f32_16x16x32_bf16(false, a, false, b,
                                                 (short)0, c, false, false);
}

// A-matrix 16x32 bf16 fragment (contiguous K, matches B loaders):
//  lane L (L<16): row M=L, VGPR v -> K=2v..2v+1
//  lane L+16    : row M=L, K ranges shifted by +16.
__device__ __forceinline__ v16bf load_a_frag(const unsigned short* src,
                                             int ld, int k0, int lane) {
  const int m = lane & 15, half = lane >> 4;
  v8u u;
#pragma unroll
  for (int v = 0; v < 8; ++v) {
    int kb = (half << 4) + (v << 1); // even
    u[v] = *(const unsigned int*)(src + m * ld + k0 + kb);
  }
  return __builtin_bit_cast(v16bf, u);
}

// B-matrix 32x16 bf16 fragment from a K-pair-packed dword array:
//  pk[(k/2)*ldw + n] holds bf16(k) | bf16(k+1)<<16.
//  lane L: col N=L&15; VGPR v: K = (L<16 ? 2v : 16+2v).
__device__ __forceinline__ v16bf load_b_frag_pk(const unsigned int* pk,
                                                int ldw, int k0, int n0,
                                                int lane) {
  const int n = lane & 15, half = lane >> 4;
  v8u u;
#pragma unroll
  for (int v = 0; v < 8; ++v) {
    int k = k0 + (half << 4) + (v << 1);
    u[v] = pk[(k >> 1) * ldw + n0 + n];
  }
  return __builtin_bit_cast(v16bf, u);
}

// B fragment where K is contiguous in memory (scores: B[k,n] = Hs[s0+n][k]).
__device__ __forceinline__ v16bf load_b_frag_kcontig(const unsigned short* src,
                                                     int ld, int k0, int lane) {
  const int n = lane & 15, half = lane >> 4;
  v8u u;
#pragma unroll
  for (int v = 0; v < 8; ++v) {
    int k = k0 + (half << 4) + (v << 1); // even
    u[v] = *(const unsigned int*)(src + n * ld + k);
  }
  return __builtin_bit_cast(v16bf, u);
}

// B fragment with K strided by ld (context: B[k,n] = Hs[k][n0+n]).
__device__ __forceinline__ v16bf load_b_frag_kstride(const unsigned short* src,
                                                     int ld, int k0, int lane) {
  const int n = lane & 15, half = lane >> 4;
  v8u u;
#pragma unroll
  for (int v = 0; v < 8; ++v) {
    int k = k0 + (half << 4) + (v << 1);
    unsigned lo = src[(size_t)k * ld + n];
    unsigned hi = src[(size_t)(k + 1) * ld + n];
    u[v] = lo | (hi << 16);
  }
  return __builtin_bit_cast(v16bf, u);
}

// ---------------------------------------------------------------------------
// Pre-pack kernels: fp32 -> bf16 K-pair-packed dwords (coalesced B-frag loads)
// ---------------------------------------------------------------------------
__global__ void pack_whh_kernel(const float* __restrict__ Whh,
                                unsigned int* __restrict__ WhhPK) {
  int i = blockIdx.x * blockDim.x + threadIdx.x;      // [512][1024] words
  if (i >= (HIDN / 2) * HIDN) return;
  int kp = i >> 10, n = i & (HIDN - 1);
  unsigned lo = f2bf(Whh[(2 * kp) * HIDN + n]);
  unsigned hi = f2bf(Whh[(2 * kp + 1) * HIDN + n]);
  WhhPK[i] = lo | (hi << 16);
}

__global__ void pack_fcw_kernel(const float* __restrict__ fc_w,
                                unsigned int* __restrict__ fcwPK) {
  int i = blockIdx.x * blockDim.x + threadIdx.x;      // [1024][128] words
  if (i >= HIDN * VOCABN) return;
  int kp = i >> 7, v = i & (VOCABN - 1);
  unsigned lo = f2bf(fc_w[v * (2 * HIDN) + 2 * kp]);      // fc_w is [V][2H]
  unsigned hi = f2bf(fc_w[v * (2 * HIDN) + 2 * kp + 1]);
  fcwPK[i] = lo | (hi << 16);
}

// ---------------------------------------------------------------------------
// RNN: 8 workgroups x 16 waves; each WG owns 16 batch rows, h kept bf16 in LDS.
// Per step: [16x1024] x [1024x1024] GEMM (wave w owns output cols w*64..+64),
// fused gather(Wxh)+bias+branchless tanh epilogue.  Whh streams from L2.
// ---------------------------------------------------------------------------
__global__ __launch_bounds__(512) void rnn_kernel(
    const int* __restrict__ x, const float* __restrict__ Wxh,
    const unsigned int* __restrict__ WhhPK, const float* __restrict__ bh,
    unsigned short* __restrict__ HsBf) {
  __shared__ unsigned short hCur[16 * HIDN]; // 32 KB
  const int tid = threadIdx.x;
  const int lane = tid & 31, w = tid >> 5;
  const int b0 = blockIdx.x * 16;
  const int m = lane & 15, half = lane >> 4;

  for (int i = tid; i < 16 * HIDN / 2; i += 512) ((unsigned int*)hCur)[i] = 0u;
  __syncthreads();

  for (int t = 0; t < TN; ++t) {
    v8f acc[4] = {};
    for (int kk = 0; kk < HIDN / 32; ++kk) {
      const int k0 = kk * 32;
      v16bf a = load_a_frag(hCur, HIDN, k0, lane);
#pragma unroll
      for (int j = 0; j < 4; ++j) {
        v16bf bb = load_b_frag_pk(WhhPK, HIDN, k0, w * 64 + j * 16, lane);
        acc[j] = wmma_bf16(a, bb, acc[j]);
      }
    }
    // Token indices for this lane's 8 C-rows: issue before the barrier so
    // the gather latency overlaps the accumulator drain.
    int xidx[8];
#pragma unroll
    for (int r = 0; r < 8; ++r)
      xidx[r] = x[(b0 + r + (half << 3)) * TN + t];
    __syncthreads(); // all reads of hCur for this step done
#pragma unroll
    for (int j = 0; j < 4; ++j) {
      const int col = w * 64 + j * 16 + m;
      const float bias = bh[col];
#pragma unroll
      for (int r = 0; r < 8; ++r) {
        const int Mr = r + (half << 3);        // C-layout: VGPR r, lane half
        const float val = acc[j][r] + Wxh[xidx[r] * HIDN + col] + bias;
        const unsigned short hb = f2bf(fast_tanh(val));
        hCur[Mr * HIDN + col] = hb;
        HsBf[(size_t)((b0 + Mr) * TN + t) * HIDN + col] = hb;
      }
    }
    __syncthreads(); // h_t fully written before next step reads it
  }
}

// ---------------------------------------------------------------------------
// Fused attention + FC: one WG per (b, 16-row t-tile).  8 waves / 256 thr.
// LDS: 32KB score strip (f32) aliased later as bf16 context tile, + 16KB P.
// ---------------------------------------------------------------------------
__global__ __launch_bounds__(256) void attn_fc_kernel(
    const unsigned short* __restrict__ HsBf,
    const unsigned int* __restrict__ fcwPK, const float* __restrict__ fc_b,
    float* __restrict__ out) {
  __shared__ __align__(16) unsigned char smem[16 * TN * 4]; // 32 KB
  __shared__ unsigned short pT[16 * TN];                    // 16 KB
  __shared__ float red[16 * 16];
  __shared__ float rowStat[16];
  float* sc = (float*)smem;                   // 16 x 512 scores
  unsigned short* cT = (unsigned short*)smem; // later: 16 x 1024 bf16 context

  const int tid = threadIdx.x, lane = tid & 31, w = tid >> 5; // w in 0..7
  const int b = blockIdx.x >> 5, tt = blockIdx.x & 31, t0 = tt * 16;
  const unsigned short* HsB = HsBf + (size_t)b * TN * HIDN;
  const unsigned short* Qrows = HsB + (size_t)t0 * HIDN;
  const int m = lane & 15, half = lane >> 4;

  // ---- scores = Q . Hs^T  (each wave: 4 s-tiles) ----
  for (int si = 0; si < 4; ++si) {
    const int s0 = (w + si * 8) * 16;
    v8f acc = {};
    for (int kk = 0; kk < HIDN / 32; ++kk) {
      const int k0 = kk * 32;
      v16bf a  = load_a_frag(Qrows, HIDN, k0, lane);
      v16bf bb = load_b_frag_kcontig(HsB + (size_t)s0 * HIDN, HIDN, k0, lane);
      acc = wmma_bf16(a, bb, acc);
    }
#pragma unroll
    for (int r = 0; r < 8; ++r) sc[(r + (half << 3)) * TN + s0 + m] = acc[r];
  }
  __syncthreads();

  // ---- softmax over 512 (16 rows x 16 slots x 32 cols) ----
  const int rr = tid >> 4, slot = tid & 15;
  {
    float mx = -1e30f;
    for (int c = slot * 32; c < slot * 32 + 32; ++c)
      mx = fmaxf(mx, sc[rr * TN + c]);
    red[rr * 16 + slot] = mx;
  }
  __syncthreads();
  if (slot == 0) {
    float mx = red[rr * 16];
    for (int i = 1; i < 16; ++i) mx = fmaxf(mx, red[rr * 16 + i]);
    rowStat[rr] = mx;
  }
  __syncthreads();
  {
    const float mxv = rowStat[rr];
    float s = 0.f;
    for (int c = slot * 32; c < slot * 32 + 32; ++c) {
      float e = __expf(sc[rr * TN + c] - mxv);
      sc[rr * TN + c] = e;
      s += e;
    }
    red[rr * 16 + slot] = s;
  }
  __syncthreads();
  if (slot == 0) {
    float s = 0.f;
    for (int i = 0; i < 16; ++i) s += red[rr * 16 + i];
    rowStat[rr] = 1.0f / s;
  }
  __syncthreads();
  {
    const float inv = rowStat[rr];
    for (int c = slot * 32; c < slot * 32 + 32; ++c)
      pT[rr * TN + c] = f2bf(sc[rr * TN + c] * inv);
  }
  __syncthreads(); // P ready; sc is dead -> its LDS becomes cT

  // ---- context = P . Hs  (each wave: 8 n-tiles) ----
  for (int ni = 0; ni < 8; ++ni) {
    const int n0 = (w + ni * 8) * 16;
    v8f acc = {};
    for (int ks = 0; ks < TN / 32; ++ks) {
      const int k0 = ks * 32;
      v16bf a  = load_a_frag(pT, TN, k0, lane);
      v16bf bb = load_b_frag_kstride(HsB + n0, HIDN, k0, lane);
      acc = wmma_bf16(a, bb, acc);
    }
#pragma unroll
    for (int r = 0; r < 8; ++r)
      cT[(r + (half << 3)) * HIDN + n0 + m] = f2bf(acc[r]);
  }
  __syncthreads();

  // ---- out = [Q | ctx] . fc_w^T + fc_b  (one 16-col v-tile per wave) ----
  {
    const int v0 = w * 16;
    v8f acc = {};
    for (int kk = 0; kk < (2 * HIDN) / 32; ++kk) {
      const int k0 = kk * 32;
      v16bf a = (k0 < HIDN) ? load_a_frag(Qrows, HIDN, k0, lane)
                            : load_a_frag(cT, HIDN, k0 - HIDN, lane);
      v16bf bb = load_b_frag_pk(fcwPK, VOCABN, k0, v0, lane);
      acc = wmma_bf16(a, bb, acc);
    }
    const float bias = fc_b[v0 + m];
#pragma unroll
    for (int r = 0; r < 8; ++r) {
      const int Mr = r + (half << 3);
      out[(size_t)(b * TN + t0 + Mr) * VOCABN + v0 + m] = acc[r] + bias;
    }
  }
}

// ---------------------------------------------------------------------------
extern "C" void kernel_launch(void* const* d_in, const int* in_sizes, int n_in,
                              void* d_out, int out_size, void* d_ws,
                              size_t ws_size, hipStream_t stream) {
  (void)in_sizes; (void)n_in; (void)out_size; (void)ws_size;
  const int*   x    = (const int*)d_in[0];
  const float* Wxh  = (const float*)d_in[1];
  const float* Whh  = (const float*)d_in[2];
  const float* bh   = (const float*)d_in[3];
  const float* fc_w = (const float*)d_in[4];
  const float* fc_b = (const float*)d_in[5];

  unsigned char* ws = (unsigned char*)d_ws;
  unsigned int*   WhhPK = (unsigned int*)ws;                       // 2 MB
  unsigned int*   fcwPK = (unsigned int*)(ws + (2u << 20));        // 512 KB
  unsigned short* HsBf  = (unsigned short*)(ws + (2u << 20) + (512u << 10)); // 128 MB

  pack_whh_kernel<<<(HIDN / 2) * HIDN / 256, 256, 0, stream>>>(Whh, WhhPK);
  pack_fcw_kernel<<<HIDN * VOCABN / 256, 256, 0, stream>>>(fc_w, fcwPK);
  rnn_kernel<<<BN / 16, 512, 0, stream>>>(x, Wxh, WhhPK, bh, HsBf);
  attn_fc_kernel<<<BN * (TN / 16), 256, 0, stream>>>(HsBf, fcwPK, fc_b,
                                                     (float*)d_out);
}
